// PartialCrossAttention_63513976373690
// MI455X (gfx1250) — compile-verified, matrix operand order checked by reading, not run on hardware
//
#include <hip/hip_runtime.h>
#include <hip/hip_bf16.h>

// ---- problem constants ----
#define S_TOT 4096
#define A_ST  16
#define B_ST  16
#define Hn    8
#define Dm    256
#define Tt    32       // A_ST + B_ST
#define HD    2048     // Hn * Dm
#define STILE 8        // s positions per block

typedef __attribute__((ext_vector_type(16))) __bf16 v16bf;
typedef __attribute__((ext_vector_type(8)))  float  v8f;
typedef __attribute__((ext_vector_type(4)))  unsigned int u32x4;
typedef __attribute__((ext_vector_type(8)))  int i32x8;
typedef __attribute__((ext_vector_type(4)))  int i32x4;

// ---- LDS layout (bf16 element offsets) ----
#define OFF_XS 0        // x tile     [STILE][Tt][Dm]          = 65536
#define OFF_QC 65536    // q chunk    [STILE][16][64]          =  8192
#define OFF_KC 73728    // k chunk    [STILE][32][64]          = 16384
#define OFF_V1 65536    // v1 chunk^T [STILE][64][32] (reuse)  = 16384
#define OFF_VC 81920    // v  chunk^T [STILE][64][32]          = 16384
#define OFF_PR 98304    // probs      [STILE][16][32]          =  4096
#define LDS_ELEMS 102400
#define LDS_BYTES (LDS_ELEMS * 2)

// ---- workspace layout (bf16 element offsets into d_ws) ----
#define WS_X   0UL                      // [S][Tt][Dm]   = 33554432 elems
#define WS_QW  33554432UL               // [16][HD][Dm]  =  8388608
#define WS_KAW 41943040UL               // [HD][Dm]      =   524288
#define WS_KBW 42467328UL               // [HD][Dm]      =   524288
#define WS_VAW 42991616UL               // [Dm][Dm]      =    65536
#define WS_VBW 43057152UL               // [Dm][Dm]      =    65536
#define WS_V2W 43122688UL               // [Hn*Tt][Tt]   =     8192
#define WS_ELEMS 43130880UL             // * 2 bytes = ~82 MB

static __device__ __forceinline__ v16bf ld_frag(const __bf16* p) {
  v16bf f;
#pragma unroll
  for (int e = 0; e < 16; ++e) f[e] = p[e];
  return f;
}

static __device__ __forceinline__ v8f wmma_bf16(v16bf a, v16bf b, v8f c) {
  return __builtin_amdgcn_wmma_f32_16x16x32_bf16(
      /*neg_a=*/false, a, /*neg_b=*/false, b,
      /*c_mod=*/(short)0, c, /*reuse_a=*/false, /*reuse_b=*/false);
}

// TDM: copy one contiguous 8192x8 bf16 tile (128KB) global -> LDS.
static __device__ __forceinline__ void tdm_load_x_tile(unsigned lds_byte,
                                                       unsigned long long gaddr) {
  u32x4 g0;
  g0[0] = 1u;                                           // count=1
  g0[1] = lds_byte;                                     // lds_addr
  g0[2] = (unsigned)(gaddr & 0xffffffffu);              // global_addr[31:0]
  g0[3] = (unsigned)((gaddr >> 32) & 0x01ffffffu) | (2u << 30);  // addr[56:32], type=2
  i32x8 g1;
  g1[0] = 0x00010000;            // workgroup_mask=0, data_size=1 (2 bytes)
  g1[1] = (int)(0x2000u << 16);  // tensor_dim0[15:0] = 8192   (bits 63:48)
  g1[2] = (int)(8u << 16);       // tensor_dim0[31:16]=0, tensor_dim1[15:0]=8
  g1[3] = (int)(0x2000u << 16);  // tensor_dim1[31:16]=0, tile_dim0 = 8192
  g1[4] = 8;                     // tile_dim1 = 8, tile_dim2 = 0
  g1[5] = 0x2000;                // tensor_dim0_stride = 8192 (low)
  g1[6] = 0;
  g1[7] = 0;
  i32x4 z4 = {0, 0, 0, 0};
#if defined(__clang_major__) && (__clang_major__ >= 23)
  i32x8 z8 = {0, 0, 0, 0, 0, 0, 0, 0};
  __builtin_amdgcn_tensor_load_to_lds(g0, g1, z4, z4, z8, 0);
#else
  __builtin_amdgcn_tensor_load_to_lds(g0, g1, z4, z4, 0);
#endif
}

// ---- prep kernel 1: pack concat(a,b) -> bf16 x[s][t][d] ----
__global__ __launch_bounds__(256)
void build_x(const float* __restrict__ a, const float* __restrict__ b,
             __bf16* __restrict__ x) {
  long i = (long)blockIdx.x * blockDim.x + threadIdx.x;   // one float4 each
  long base = i * 4;
  int d = (int)(base & 255);
  int t = (int)((base >> 8) & 31);
  long s = base >> 13;
  const float* src = (t < A_ST) ? &a[(s * A_ST + t) * Dm + d]
                                : &b[(s * B_ST + (t - A_ST)) * Dm + d];
  float4 v = *(const float4*)src;
  __bf16* dst = &x[base];
  dst[0] = (__bf16)v.x; dst[1] = (__bf16)v.y;
  dst[2] = (__bf16)v.z; dst[3] = (__bf16)v.w;
}

// ---- prep kernel 2: generic fp32 -> bf16 (n multiple of 4) ----
__global__ __launch_bounds__(256)
void cvt_w(const float* __restrict__ src, __bf16* __restrict__ dst, long n4) {
  long i = (long)blockIdx.x * blockDim.x + threadIdx.x;
  if (i >= n4) return;
  float4 v = ((const float4*)src)[i];
  __bf16* d = dst + i * 4;
  d[0] = (__bf16)v.x; d[1] = (__bf16)v.y;
  d[2] = (__bf16)v.z; d[3] = (__bf16)v.w;
}

// ---- main fused kernel ----
__global__ __launch_bounds__(256, 1)
void pca_fused(const __bf16* __restrict__ xab, const __bf16* __restrict__ qw,
               const __bf16* __restrict__ kaw, const __bf16* __restrict__ kbw,
               const __bf16* __restrict__ vaw, const __bf16* __restrict__ vbw,
               const __bf16* __restrict__ v2w,
               const float* __restrict__ qb,  const float* __restrict__ kab,
               const float* __restrict__ kbb, const float* __restrict__ vab,
               const float* __restrict__ vbb, float* __restrict__ out) {
  extern __shared__ char smem_raw[];
  __bf16* sm = (__bf16*)smem_raw;

  const int s0   = blockIdx.x * STILE;
  const int h    = blockIdx.y;
  const int tid  = threadIdx.x;
  const int lane = tid & 31;
  const int wave = tid >> 5;
  const int lo   = lane & 15;
  const int hi   = lane >> 4;

  const __bf16* kwg[2] = {kaw, kbw};
  const float*  kbg[2] = {kab, kbb};
  const __bf16* vwg[2] = {vaw, vbw};
  const float*  vbg[2] = {vab, vbb};

  // ---- Phase 1: TDM copy of the bf16 x tile (contiguous 128KB) into LDS ----
  if (wave == 0) {
    const __bf16* xsrc = xab + (long)s0 * Tt * Dm;
    unsigned lds_byte = (unsigned)(uintptr_t)(&sm[OFF_XS]);
    tdm_load_x_tile(lds_byte, (unsigned long long)(uintptr_t)xsrc);
    __builtin_amdgcn_s_wait_tensorcnt((short)0);
  }
  __syncthreads();

  // persistent per-wave score accumulators: wave owns ss = wave
  v8f sc0 = {};  // keys j = 0..15
  v8f sc1 = {};  // keys j = 16..31

  // ---- Phase 2: Q/K projection chunks + score accumulation over D ----
  for (int dc = 0; dc < 4; ++dc) {
    // 2a: q chunk: per-stream weights
    for (int tile = wave; tile < 64; tile += 8) {
      int n  = tile >> 2;
      int ct = tile & 3;
      int ocol = h * Dm + dc * 64 + ct * 16 + lo;
      __builtin_prefetch(&qw[((long)n * HD + ocol) * Dm], 0, 0);
      float bias = qb[n * HD + ocol];
      v8f c = {bias, bias, bias, bias, bias, bias, bias, bias};
#pragma unroll
      for (int kk = 0; kk < 8; ++kk) {
        int ss = lo & 7;
        int k  = kk * 32 + hi * 16;
        v16bf af  = ld_frag(&sm[OFF_XS + (ss * Tt + n) * Dm + k]);
        v16bf bfw = ld_frag(&qw[((long)n * HD + ocol) * Dm + k]);
        c = wmma_bf16(af, bfw, c);
      }
#pragma unroll
      for (int r = 0; r < 8; ++r) {
        int m = r + hi * 8;
        if (m < STILE)
          sm[OFF_QC + (m * 16 + n) * 64 + ct * 16 + lo] = (__bf16)c[r];
      }
    }
    // 2b: k chunk: shared weights per stream group
    for (int tile = wave; tile < 64; tile += 8) {
      int ss = tile >> 3;
      int g  = (tile >> 2) & 1;
      int ct = tile & 3;
      int ocol = h * Dm + dc * 64 + ct * 16 + lo;
      const __bf16* W = kwg[g];
      float bias = kbg[g][ocol];
      v8f c = {bias, bias, bias, bias, bias, bias, bias, bias};
#pragma unroll
      for (int kk = 0; kk < 8; ++kk) {
        int t = g * 16 + lo;
        int k = kk * 32 + hi * 16;
        v16bf af  = ld_frag(&sm[OFF_XS + (ss * Tt + t) * Dm + k]);
        v16bf bfw = ld_frag(&W[(long)ocol * Dm + k]);
        c = wmma_bf16(af, bfw, c);
      }
#pragma unroll
      for (int r = 0; r < 8; ++r) {
        int t = g * 16 + r + hi * 8;
        sm[OFF_KC + (ss * Tt + t) * 64 + ct * 16 + lo] = (__bf16)c[r];
      }
    }
    __syncthreads();
    // 2c: scores += q_chunk . k_chunk^T
    {
      int ss = wave;
#pragma unroll
      for (int kk = 0; kk < 2; ++kk) {
        int k = kk * 32 + hi * 16;
        v16bf aq = ld_frag(&sm[OFF_QC + (ss * 16 + lo) * 64 + k]);
        v16bf b0 = ld_frag(&sm[OFF_KC + (ss * Tt + lo) * 64 + k]);
        v16bf b1 = ld_frag(&sm[OFF_KC + (ss * Tt + 16 + lo) * 64 + k]);
        sc0 = wmma_bf16(aq, b0, sc0);
        sc1 = wmma_bf16(aq, b1, sc1);
      }
    }
    __syncthreads();
  }

  // ---- Phase 3: softmax over the 32 keys, probs -> LDS bf16 ----
  {
    int ss = wave;
    const float scl = 0.0625f;                 // 1/sqrt(256)
#pragma unroll
    for (int r = 0; r < 8; ++r) {
      float x0 = sc0[r] * scl;
      float x1 = sc1[r] * scl;
      float mx = fmaxf(x0, x1);
      for (int o = 1; o < 16; o <<= 1) mx = fmaxf(mx, __shfl_xor(mx, o, 16));
      float e0 = __expf(x0 - mx);
      float e1 = __expf(x1 - mx);
      float sum = e0 + e1;
      for (int o = 1; o < 16; o <<= 1) sum += __shfl_xor(sum, o, 16);
      float inv = 1.0f / sum;
      int i = r + hi * 8;
      sm[OFF_PR + (ss * 16 + i) * Tt + lo]      = (__bf16)(e0 * inv);
      sm[OFF_PR + (ss * 16 + i) * Tt + 16 + lo] = (__bf16)(e1 * inv);
    }
  }
  __syncthreads();

  // ---- Phase 4: v1 -> v (stream mix) -> out, in 64-wide D chunks ----
  for (int dc = 0; dc < 4; ++dc) {
    // 4a: v1 chunk (stored transposed [dd][t])
    for (int tile = wave; tile < 64; tile += 8) {
      int ss = tile >> 3;
      int g  = (tile >> 2) & 1;
      int ct = tile & 3;
      int o  = dc * 64 + ct * 16 + lo;
      const __bf16* W = vwg[g];
      float bias = vbg[g][o];
      v8f c = {bias, bias, bias, bias, bias, bias, bias, bias};
#pragma unroll
      for (int kk = 0; kk < 8; ++kk) {
        int t = g * 16 + lo;
        int k = kk * 32 + hi * 16;
        v16bf af  = ld_frag(&sm[OFF_XS + (ss * Tt + t) * Dm + k]);
        v16bf bfw = ld_frag(&W[(long)o * Dm + k]);
        c = wmma_bf16(af, bfw, c);
      }
#pragma unroll
      for (int r = 0; r < 8; ++r) {
        int t  = g * 16 + r + hi * 8;
        int dd = ct * 16 + lo;
        sm[OFF_V1 + (ss * 64 + dd) * Tt + t] = (__bf16)c[r];
      }
    }
    __syncthreads();
    // 4b: v chunk = v2_w[h-slice] @ v1 chunk (K = 32)
    for (int tile = wave; tile < 64; tile += 8) {
      int ss = tile >> 3;
      int jt = (tile >> 2) & 1;
      int ct = tile & 3;
      int j  = jt * 16 + lo;
      v16bf af  = ld_frag(&v2w[(h * Tt + j) * Tt + hi * 16]);
      v16bf bfw = ld_frag(&sm[OFF_V1 + (ss * 64 + ct * 16 + lo) * Tt + hi * 16]);
      v8f c = {};
      c = wmma_bf16(af, bfw, c);
#pragma unroll
      for (int r = 0; r < 8; ++r) {
        int jr = jt * 16 + r + hi * 8;
        int dd = ct * 16 + lo;
        sm[OFF_VC + (ss * 64 + dd) * Tt + jr] = (__bf16)c[r];
      }
    }
    __syncthreads();
    // 4c: out chunk = probs @ v chunk (K = 32), fp32 stores
    {
      int ss = wave;
      v16bf ap = ld_frag(&sm[OFF_PR + (ss * 16 + lo) * Tt + hi * 16]);
#pragma unroll
      for (int ct = 0; ct < 4; ++ct) {
        v16bf bfw = ld_frag(&sm[OFF_VC + (ss * 64 + ct * 16 + lo) * Tt + hi * 16]);
        v8f c = {};
        c = wmma_bf16(ap, bfw, c);
        int dd = dc * 64 + ct * 16 + lo;
#pragma unroll
        for (int r = 0; r < 8; ++r) {
          int i = r + hi * 8;
          out[(((long)(s0 + ss) * A_ST + i) * Hn + h) * Dm + dd] = c[r];
        }
      }
    }
    __syncthreads();
  }
}

extern "C" void kernel_launch(void* const* d_in, const int* in_sizes, int n_in,
                              void* d_out, int out_size, void* d_ws, size_t ws_size,
                              hipStream_t stream) {
  (void)in_sizes; (void)n_in; (void)out_size; (void)ws_size;
  const float* a   = (const float*)d_in[0];
  const float* b   = (const float*)d_in[1];
  const float* qw  = (const float*)d_in[2];
  const float* qb  = (const float*)d_in[3];
  const float* kaw = (const float*)d_in[4];
  const float* kab = (const float*)d_in[5];
  const float* kbw = (const float*)d_in[6];
  const float* kbb = (const float*)d_in[7];
  const float* vaw = (const float*)d_in[8];
  const float* vab = (const float*)d_in[9];
  const float* vbw = (const float*)d_in[10];
  const float* vbb = (const float*)d_in[11];
  const float* v2w = (const float*)d_in[12];
  float* out = (float*)d_out;
  __bf16* ws = (__bf16*)d_ws;

  // ---- prep: bf16 conversions into workspace ----
  {
    long nx4 = (long)S_TOT * Tt * Dm / 4;                    // 8388608
    build_x<<<(unsigned)(nx4 / 256), 256, 0, stream>>>(a, b, ws + WS_X);
    struct { const float* s; unsigned long off; long n; } cv[6] = {
        {qw,  WS_QW,  (long)A_ST * HD * Dm},
        {kaw, WS_KAW, (long)HD * Dm},
        {kbw, WS_KBW, (long)HD * Dm},
        {vaw, WS_VAW, (long)Dm * Dm},
        {vbw, WS_VBW, (long)Dm * Dm},
        {v2w, WS_V2W, (long)Hn * Tt * Tt}};
    for (int i = 0; i < 6; ++i) {
      long n4 = cv[i].n / 4;
      unsigned blocks = (unsigned)((n4 + 255) / 256);
      cvt_w<<<blocks, 256, 0, stream>>>(cv[i].s, ws + cv[i].off, n4);
    }
  }

  (void)hipFuncSetAttribute((const void*)pca_fused,
                            hipFuncAttributeMaxDynamicSharedMemorySize,
                            LDS_BYTES);
  dim3 grid(S_TOT / STILE, Hn);
  pca_fused<<<grid, 256, LDS_BYTES, stream>>>(
      ws + WS_X, ws + WS_QW, ws + WS_KAW, ws + WS_KBW, ws + WS_VAW,
      ws + WS_VBW, ws + WS_V2W, qb, kab, kbb, vab, vbb, out);
}
